// DQSN_1924145348759
// MI455X (gfx1250) — matrix operand order repaired and checked by reading, not compiled, not run on hardware
//
#include <hip/hip_runtime.h>

// Fused spiking-MLP forward for MI455X (gfx1250, wave32, WMMA), v6 (= v4 cfg).
//
// Z = W @ S^T orientation: weights = WMMA A operand (LDS fp16, rows permuted
// by sigma = swap bits 3<->4), spikes = B operand built purely in registers
// (C/D layout of layer l == B layout of layer l+1 under sigma). Membrane
// potentials are the WMMA accumulators; loop-invariant drive c0 parked in
// per-wave LDS (f32). __launch_bounds__(128, 1): the allocator needs ~770
// VGPRs to be spill-free (512-cap configs spill ~260 ops into the hot loop);
// spill-free single-wave-per-SIMD with 8 independent WMMA chains wins.

typedef __attribute__((ext_vector_type(16))) _Float16 v16h;
typedef __attribute__((ext_vector_type(8)))  _Float16 v8h;
typedef __attribute__((ext_vector_type(4)))  _Float16 v4h;
typedef __attribute__((ext_vector_type(8)))  float    v8f;
typedef __attribute__((ext_vector_type(4)))  float    v4f;

#define T_STEPS 16

// ---- LDS byte layout ----
#define OFF_W0   0        // 128x128 fp16 = 32768 B
#define OFF_W1   32768
#define OFF_W2   65536
#define OFF_W3   98304    // 64x128 fp16 = 16384 B
#define OFF_B0   114688   // 128 f32 (sigma-permuted)
#define OFF_B1   115200
#define OFF_B2   115712
#define OFF_B3   116224   // 64 f32
#define OFF_CH   116480   // 4 waves x 16x128 f32 c0 tiles (8192 B each)
#define SMEM_BYTES (116480 + 4 * 8192)   // 149248 B

// sigma: swap bits 3 and 4 (self-inverse); staging-time feature permutation.
__device__ __forceinline__ int sigma(int r) {
  return (r & ~24) | ((r & 8) << 1) | ((r & 16) >> 1);
}

// A-fragment (16x32 fp16) from row-major [*,128] fp16 LDS weights, contiguous K order (not ISA table order):
// lane row = lane&15; K = kc*32 + 16*hi + {0..15}. Two ds_load_b128.
__device__ __forceinline__ v16h ld_fragA(const _Float16* W, int row, int kc, int lane) {
  const int off = (lane >> 4) << 4;
  const _Float16* p = W + row * 128 + kc * 32 + off;
  v8h lo = *(const v8h*)(p);
  v8h hi = *(const v8h*)(p + 8);
  return __builtin_shufflevector(lo, hi, 0,1,2,3,4,5,6,7,8,9,10,11,12,13,14,15);
}

// IF fire + hard reset on 8 C/D tiles, writing the next GEMM's 4 B-fragments
// directly (tile 2kc -> halves 0..7, tile 2kc+1 -> halves 8..15). All in-lane.
__device__ __forceinline__ void fire_build(v8f* v, v16h* bf) {
#pragma unroll
  for (int kc = 0; kc < 4; ++kc) {
    v16h f;
#pragma unroll
    for (int b = 0; b < 2; ++b) {
#pragma unroll
      for (int j = 0; j < 8; ++j) {
        const int nt = 2 * kc + b;
        const float val = v[nt][j];
        const bool s = (val >= 1.0f);              // VTH = 1.0
        f[8 * b + j] = s ? (_Float16)1.0f : (_Float16)0.0f;
        v[nt][j] = s ? 0.0f : val;                 // hard reset
      }
    }
    bf[kc] = f;
  }
}

// acc[NT] += W'(A, sigma-ordered rows) @ B. kc-outer / nt-inner: consecutive
// WMMAs hit different accumulators (hides D->C RAW latency); B reused NT x.
template<int NT>
__device__ __forceinline__ void gemm_accB(const _Float16* W, const v16h* bf,
                                          int lane, v8f* acc) {
  const int nloc = lane & 15;
#pragma unroll
  for (int kc = 0; kc < 4; ++kc) {
#pragma unroll
    for (int nt = 0; nt < NT; ++nt) {
      v16h af = ld_fragA(W, nt * 16 + nloc, kc, lane);
      acc[nt] = __builtin_amdgcn_wmma_f32_16x16x32_f16(
          false, af, false, bf[kc], (short)0, acc[nt], false, false);
    }
  }
}

__global__ __launch_bounds__(128, 1)
void snn_fused_kernel(const float* __restrict__ x,
                      const float* __restrict__ W0, const float* __restrict__ b0,
                      const float* __restrict__ W1, const float* __restrict__ b1,
                      const float* __restrict__ W2, const float* __restrict__ b2,
                      const float* __restrict__ W3, const float* __restrict__ b3,
                      float* __restrict__ out) {
  extern __shared__ char smem[];
  _Float16* w0h = (_Float16*)(smem + OFF_W0);
  _Float16* w1h = (_Float16*)(smem + OFF_W1);
  _Float16* w2h = (_Float16*)(smem + OFF_W2);
  _Float16* w3h = (_Float16*)(smem + OFF_W3);
  float* b0f = (float*)(smem + OFF_B0);
  float* b1f = (float*)(smem + OFF_B1);
  float* b2f = (float*)(smem + OFF_B2);
  float* b3f = (float*)(smem + OFF_B3);

  const int tid  = threadIdx.x;                     // 0..127 (4 waves)
  const int wave = tid >> 5;
  const int lane = tid & 31;
  const int hi   = lane >> 4;
  const int m    = lane & 15;                       // batch column within tile

  // ---- stage weights fp32 -> fp16 LDS, rows permuted by sigma ----
#pragma unroll 4
  for (int i = tid; i < 128 * 128 / 4; i += 128) {
    const int r = i >> 5;
    const int k4 = i & 31;
    const int src = sigma(r) * 32 + k4;
    *(v4h*)(w0h + r * 128 + k4 * 4) = __builtin_convertvector(((const v4f*)W0)[src], v4h);
    *(v4h*)(w1h + r * 128 + k4 * 4) = __builtin_convertvector(((const v4f*)W1)[src], v4h);
    *(v4h*)(w2h + r * 128 + k4 * 4) = __builtin_convertvector(((const v4f*)W2)[src], v4h);
  }
#pragma unroll 4
  for (int i = tid; i < 64 * 128 / 4; i += 128) {
    const int r = i >> 5;
    const int k4 = i & 31;
    *(v4h*)(w3h + r * 128 + k4 * 4) =
        __builtin_convertvector(((const v4f*)W3)[sigma(r) * 32 + k4], v4h);
  }
  if (tid < 128) { b0f[tid] = b0[sigma(tid)]; b1f[tid] = b1[sigma(tid)];
                   b2f[tid] = b2[sigma(tid)]; }
  if (tid < 64)  { b3f[tid] = b3[sigma(tid)]; }
  __syncthreads();

  const int row0 = blockIdx.x * 64 + wave * 16;     // this wave's 16 batch rows
  float* chw = (float*)(smem + OFF_CH) + wave * (16 * 128);  // per-wave c0 tile

  // ---- B-fragments of x^T straight from global (lane = batch col m,
  //      K = kc*32 + 16*hi + t => 16 contiguous floats of row m) ----
  v16h xb[4];
#pragma unroll
  for (int kc = 0; kc < 4; ++kc) {
    const float* px = x + (size_t)(row0 + m) * 128 + kc * 32 + 16 * hi;
    v4h h0 = __builtin_convertvector(*(const v4f*)(px + 0),  v4h);
    v4h h1 = __builtin_convertvector(*(const v4f*)(px + 4),  v4h);
    v4h h2 = __builtin_convertvector(*(const v4f*)(px + 8),  v4h);
    v4h h3 = __builtin_convertvector(*(const v4f*)(px + 12), v4h);
    v8h lo = __builtin_shufflevector(h0, h1, 0,1,2,3,4,5,6,7);
    v8h hh = __builtin_shufflevector(h2, h3, 0,1,2,3,4,5,6,7);
    xb[kc] = __builtin_shufflevector(lo, hh, 0,1,2,3,4,5,6,7,8,9,10,11,12,13,14,15);
  }

  // ---- c0 = W0' @ x^T + b0' (loop-invariant): compute once, park in LDS ----
  {
    v8f c0[8];
#pragma unroll
    for (int nt = 0; nt < 8; ++nt) c0[nt] = *(const v8f*)(b0f + nt * 16 + 8 * hi);
    gemm_accB<8>(w0h, xb, lane, c0);
#pragma unroll
    for (int nt = 0; nt < 8; ++nt)
      *(v8f*)(chw + m * 128 + nt * 16 + 8 * hi) = c0[nt];   // wave-private, in-order DS
  }

  v8f v0[8], v1[8], v2[8], vL[4];
#pragma unroll
  for (int nt = 0; nt < 8; ++nt) { v0[nt] = (v8f)0.0f; v1[nt] = (v8f)0.0f; v2[nt] = (v8f)0.0f; }
#pragma unroll
  for (int nt = 0; nt < 4; ++nt) vL[nt] = (v8f)0.0f;

  // ---- T timesteps (rolled) ----
#pragma clang loop unroll(disable)
  for (int t = 0; t < T_STEPS; ++t) {
    v16h bf[4];

    // layer 0: charge with c0 (from LDS), fire, reset -> B-fragments
#pragma unroll
    for (int nt = 0; nt < 8; ++nt)
      v0[nt] += *(const v8f*)(chw + m * 128 + nt * 16 + 8 * hi);
    fire_build(v0, bf);

    // layer 1: v1 += b1' (pre-add, same sum); v1 += W1' @ s0; fire
#pragma unroll
    for (int nt = 0; nt < 8; ++nt) v1[nt] += *(const v8f*)(b1f + nt * 16 + 8 * hi);
    gemm_accB<8>(w1h, bf, lane, v1);
    fire_build(v1, bf);

    // layer 2
#pragma unroll
    for (int nt = 0; nt < 8; ++nt) v2[nt] += *(const v8f*)(b2f + nt * 16 + 8 * hi);
    gemm_accB<8>(w2h, bf, lane, v2);
    fire_build(v2, bf);

    // output layer folded into vL: vL = 0.5*(vL + b3' + W3' @ s2)
#pragma unroll
    for (int nt = 0; nt < 4; ++nt) vL[nt] += *(const v8f*)(b3f + nt * 16 + 8 * hi);
    gemm_accB<4>(w3h, bf, lane, vL);
#pragma unroll
    for (int nt = 0; nt < 4; ++nt) vL[nt] *= 0.5f;
  }

  // ---- store vL: per nt one contiguous v8f -> two global_store_b128 ----
#pragma unroll
  for (int nt = 0; nt < 4; ++nt) {
    const int o0 = 32 * (nt >> 1) + 16 * hi + 8 * (nt & 1);  // = sigma(nt*16+8*hi)
    *(v8f*)(out + (size_t)(row0 + m) * 64 + o0) = vL[nt];
  }
}

extern "C" void kernel_launch(void* const* d_in, const int* in_sizes, int n_in,
                              void* d_out, int out_size, void* d_ws, size_t ws_size,
                              hipStream_t stream) {
  const float* x  = (const float*)d_in[0];
  const float* W0 = (const float*)d_in[1];
  const float* b0 = (const float*)d_in[2];
  const float* W1 = (const float*)d_in[3];
  const float* b1 = (const float*)d_in[4];
  const float* W2 = (const float*)d_in[5];
  const float* b2 = (const float*)d_in[6];
  const float* W3 = (const float*)d_in[7];
  const float* b3 = (const float*)d_in[8];
  float* out = (float*)d_out;

  const int B = in_sizes[0] / 128;      // 65536
  dim3 grid(B / 64), block(128);        // 4 waves/block, 16 batch rows/wave
  snn_fused_kernel<<<grid, block, SMEM_BYTES, stream>>>(
      x, W0, b0, W1, b1, W2, b2, W3, b3, out);
}
